// VSSBlock_75754633167197
// MI455X (gfx1250) — matrix operand order repaired and checked by reading, not compiled, not run on hardware
//
#include <hip/hip_runtime.h>
#include <hip/hip_bf16.h>
#include <math.h>

// Problem constants (compile-time; shapes fixed by the reference)
#define BATCH 4
#define HH    64
#define WWD   64
#define CC    96
#define DI    192
#define NS    16
#define RR    6
#define KDIR  4
#define LSEQ  4096
#define BLROW 16384          // BATCH * LSEQ
#define PLD   48             // padded x_proj output row width

typedef __attribute__((ext_vector_type(16))) __bf16 v16bf;
typedef __attribute__((ext_vector_type(8)))  float  v8f;

union Frag {
    v16bf bf;
    uint4 q[2];
    unsigned short u[16];
};

// ---------- helpers ----------
__device__ __forceinline__ unsigned short f2bf(float f) {
    unsigned int u = __float_as_uint(f);
    unsigned int r = u + 0x7FFFu + ((u >> 16) & 1u);   // round-nearest-even
    return (unsigned short)(r >> 16);
}

__device__ __forceinline__ float wave_sum(float v) {
#pragma unroll
    for (int o = 16; o > 0; o >>= 1) v += __shfl_xor(v, o, 32);
    return v;
}

__device__ __forceinline__ float silu(float x) {
    return x / (1.0f + __expf(-x));
}

// pixel index visited at step l of direction k
__device__ __forceinline__ int pix_of(int k, int l) {
    int m = (k & 2) ? (LSEQ - 1 - l) : l;
    return (k & 1) ? (((m & 63) << 6) | (m >> 6)) : m;   // transpose for k=1,3
}

// ---------- 0. weight prep: fp32 -> bf16 staging, A = -exp(A_logs) ----------
__global__ __launch_bounds__(256) void prep_kernel(
    const float* __restrict__ in_proj_w,   // [384][96]
    const float* __restrict__ x_proj_w,    // [4][38][192]
    const float* __restrict__ out_proj_w,  // [96][192]
    const float* __restrict__ A_logs,      // [4*192][16]
    unsigned short* __restrict__ wt_in,    // [384][96] bf16
    unsigned short* __restrict__ wt_xp,    // [4][48][192] bf16 (rows remapped/padded)
    unsigned short* __restrict__ wt_out,   // [96][192] bf16
    float* __restrict__ Aexp)              // [4*192][16]
{
    int i = blockIdx.x * 256 + threadIdx.x;
    const int N_IN = 384 * 96;             // 36864
    const int N_XP = KDIR * PLD * DI;      // 36864
    const int N_OUT = 96 * 192;            // 18432
    const int N_A = KDIR * DI * NS;        // 12288
    if (i < N_IN) {
        wt_in[i] = f2bf(in_proj_w[i]);
        return;
    }
    i -= N_IN;
    if (i < N_XP) {
        int k = i / (PLD * DI);
        int rem = i % (PLD * DI);
        int np = rem / DI;                 // padded row 0..47
        int kd = rem % DI;
        // dt rows 0..5 -> 0..5 ; B rows 6..21 -> 8..23 ; C rows 22..37 -> 24..39
        int n = (np < 6) ? np : ((np >= 8 && np < 40) ? (np - 2) : -1);
        wt_xp[i] = (n >= 0) ? f2bf(x_proj_w[(k * 38 + n) * DI + kd]) : (unsigned short)0;
        return;
    }
    i -= N_XP;
    if (i < N_OUT) {
        wt_out[i] = f2bf(out_proj_w[i]);
        return;
    }
    i -= N_OUT;
    if (i < N_A) {
        Aexp[i] = -__expf(A_logs[i]);
    }
}

__global__ __launch_bounds__(256) void zero_kernel(float* __restrict__ p, int n) {
    int i = blockIdx.x * 256 + threadIdx.x;
    if (i < n) p[i] = 0.0f;
}

// ---------- 1. input layernorm (C=96) -> bf16 rows ----------
__global__ __launch_bounds__(256) void ln_in_kernel(
    const float* __restrict__ x, const float* __restrict__ g, const float* __restrict__ b,
    unsigned short* __restrict__ hnorm)
{
    int wave = threadIdx.x >> 5, lane = threadIdx.x & 31;
    int row = blockIdx.x * 8 + wave;
    if (row >= BLROW) return;
    const float* xr = x + (long)row * CC;
    float v0 = xr[lane], v1 = xr[lane + 32], v2 = xr[lane + 64];
    float s  = wave_sum(v0 + v1 + v2);
    float s2 = wave_sum(v0 * v0 + v1 * v1 + v2 * v2);
    float mu = s * (1.0f / CC);
    float var = s2 * (1.0f / CC) - mu * mu;
    float rstd = rsqrtf(var + 1e-5f);
    unsigned short* o = hnorm + (long)row * CC;
    o[lane]      = f2bf((v0 - mu) * rstd * g[lane]      + b[lane]);
    o[lane + 32] = f2bf((v1 - mu) * rstd * g[lane + 32] + b[lane + 32]);
    o[lane + 64] = f2bf((v2 - mu) * rstd * g[lane + 64] + b[lane + 64]);
}

// ---------- 2. generic bf16 WMMA GEMM: C[M][ldc] = A[M][lda] * B^T, B stored [N][ldb] ----------
__global__ __launch_bounds__(256) void wmma_gemm_nt(
    const unsigned short* __restrict__ A,
    const unsigned short* __restrict__ Bw,
    float* __restrict__ C,
    const float* __restrict__ resid,       // optional, same layout as C
    int M, int N, int K, int lda, int ldb, int ldc,
    long strideB, long strideC)            // per blockIdx.y batch
{
    const unsigned short* Bp = Bw + (long)blockIdx.y * strideB;
    float* Cp = C + (long)blockIdx.y * strideC;
    int wave = threadIdx.x >> 5;
    int lane = threadIdx.x & 31;
    int tilesN = N >> 4;
    long tile = (long)blockIdx.x * 8 + wave;
    long tot = (long)(M >> 4) * tilesN;
    if (tile >= tot) return;               // wave-uniform: EXEC stays full for WMMA
    int tm = (int)(tile / tilesN), tn = (int)(tile % tilesN);
    int m0 = tm << 4, n0 = tn << 4;
    int hi = lane >> 4;                    // lane half (0/1)
    int lr = lane & 15;
    const unsigned short* arow = A  + (long)(m0 + lr) * lda;
    const unsigned short* brow = Bp + (long)(n0 + lr) * ldb;
    v8f acc = {};
    for (int kk = 0; kk < K; kk += 32) {
        Frag a, b;
        // A 16x32 bf16 fragment: lanes 0-15 K=kk+[0..7],[16..23]; lanes 16-31 +8
        a.q[0] = *(const uint4*)(arow + kk + hi * 8);
        a.q[1] = *(const uint4*)(arow + kk + 16 + hi * 8);
        // B 32x16 fragment from [N][K] rows: lane n = n0+lr, K = kk+[0..7],[16..23]; lanes 16-31 +8
        b.q[0] = *(const uint4*)(brow + kk + hi * 8);
        b.q[1] = *(const uint4*)(brow + kk + 16 + hi * 8);
        acc = __builtin_amdgcn_wmma_f32_16x16x32_bf16(
            false, a.bf, false, b.bf, (short)0, acc, false, false);
    }
    // C/D layout: lane -> N = lr; VGPR v -> M = v + hi*8
    float* crow = Cp + (long)(m0 + hi * 8) * ldc + n0 + lr;
    if (resid) {
        const float* rrow = resid + (long)(m0 + hi * 8) * ldc + n0 + lr;
#pragma unroll
        for (int v = 0; v < 8; ++v) crow[(long)v * ldc] = acc[v] + rrow[(long)v * ldc];
    } else {
#pragma unroll
        for (int v = 0; v < 8; ++v) crow[(long)v * ldc] = acc[v];
    }
}

// ---------- 3. depthwise 3x3 conv + bias + SiLU ----------
__global__ __launch_bounds__(256) void conv_dw_silu_kernel(
    const float* __restrict__ xz,          // [BLROW][384], xc = cols 0..191
    const float* __restrict__ cw,          // [192][1][3][3]
    const float* __restrict__ cb,          // [192]
    float* __restrict__ xconv,             // [BLROW][192] fp32
    unsigned short* __restrict__ xcbf)     // [BLROW][192] bf16
{
    long idx = (long)blockIdx.x * 256 + threadIdx.x;
    if (idx >= (long)BLROW * DI) return;
    int d = (int)(idx % DI);
    long t = idx / DI;
    int pix = (int)(t % LSEQ);
    int b = (int)(t / LSEQ);
    int h = pix >> 6, w = pix & 63;
    float acc = cb[d];
#pragma unroll
    for (int ky = 0; ky < 3; ++ky) {
        int ih = h + ky - 1;
        if ((unsigned)ih >= HH) continue;
#pragma unroll
        for (int kx = 0; kx < 3; ++kx) {
            int iw = w + kx - 1;
            if ((unsigned)iw >= WWD) continue;
            acc += xz[((long)b * LSEQ + ih * WWD + iw) * 384 + d] * cw[d * 9 + ky * 3 + kx];
        }
    }
    float s = silu(acc);
    xconv[idx] = s;
    xcbf[idx] = f2bf(s);
}

// ---------- 4. selective scan: one block per (b,k), one lane per channel d ----------
__global__ __launch_bounds__(192) void scan_kernel(
    const float* __restrict__ proj,        // [K][BLROW][48] (dt:0-5, B:8-23, C:24-39)
    const float* __restrict__ xconv,       // [BLROW][192]
    const float* __restrict__ dt_w,        // [K][192][6]
    const float* __restrict__ dt_b,        // [K][192]
    const float* __restrict__ Aexp,        // [K*192][16]
    const float* __restrict__ Ds,          // [K*192]
    float* __restrict__ ysum)              // [BLROW][192] accumulated over k
{
    int b = blockIdx.x >> 2;
    int k = blockIdx.x & 3;
    int d = threadIdx.x;                   // 0..191
    int kd = k * DI + d;

    float A[NS];
#pragma unroll
    for (int n = 0; n < NS; ++n) A[n] = Aexp[(long)kd * NS + n];
    float dtw[RR];
#pragma unroll
    for (int r = 0; r < RR; ++r) dtw[r] = dt_w[(long)kd * RR + r];
    const float dtb = dt_b[kd];
    const float Dkd = Ds[kd];

    float h[NS] = {0};
    const float* projk = proj + ((long)k * BLROW + (long)b * LSEQ) * PLD;
    const float* xcb   = xconv + (long)b * LSEQ * DI;
    float* ysb         = ysum  + (long)b * LSEQ * DI;

    for (int l = 0; l < LSEQ; ++l) {
        int pix = pix_of(k, l);
        const float* p = projk + (long)pix * PLD;
        if (l + 1 < LSEQ) {
            int pn = pix_of(k, l + 1);
            __builtin_prefetch(projk + (long)pn * PLD, 0, 1);   // global_prefetch_b8
            __builtin_prefetch(xcb + (long)pn * DI + d, 0, 1);
        }
        float dtr = dtb;
#pragma unroll
        for (int r = 0; r < RR; ++r) dtr += p[r] * dtw[r];
        float dt = (dtr > 20.0f) ? dtr : log1pf(__expf(dtr));   // softplus
        float u = xcb[(long)pix * DI + d];
        float du = dt * u;
        float y = Dkd * u;
        float4 B0 = *(const float4*)(p + 8);
        float4 B1 = *(const float4*)(p + 12);
        float4 B2 = *(const float4*)(p + 16);
        float4 B3 = *(const float4*)(p + 20);
        float4 C0 = *(const float4*)(p + 24);
        float4 C1 = *(const float4*)(p + 28);
        float4 C2 = *(const float4*)(p + 32);
        float4 C3 = *(const float4*)(p + 36);
        const float* Bf = (const float*)&B0;   // contiguous float4 quads
        const float* Cf = (const float*)&C0;
        float Bv[NS] = {B0.x,B0.y,B0.z,B0.w,B1.x,B1.y,B1.z,B1.w,
                        B2.x,B2.y,B2.z,B2.w,B3.x,B3.y,B3.z,B3.w};
        float Cv[NS] = {C0.x,C0.y,C0.z,C0.w,C1.x,C1.y,C1.z,C1.w,
                        C2.x,C2.y,C2.z,C2.w,C3.x,C3.y,C3.z,C3.w};
        (void)Bf; (void)Cf;
#pragma unroll
        for (int n = 0; n < NS; ++n) {
            h[n] = __expf(dt * A[n]) * h[n] + du * Bv[n];
            y += h[n] * Cv[n];
        }
        atomicAdd(&ysb[(long)pix * DI + d], y);
    }
}

// ---------- 5. out-norm layernorm (D=192) + SiLU(z) gate -> bf16 ----------
__global__ __launch_bounds__(256) void merge_gate_kernel(
    const float* __restrict__ ysum,        // [BLROW][192]
    const float* __restrict__ g, const float* __restrict__ b,
    const float* __restrict__ xz,          // [BLROW][384], z = cols 192..383
    unsigned short* __restrict__ gated)    // [BLROW][192] bf16
{
    int wave = threadIdx.x >> 5, lane = threadIdx.x & 31;
    int row = blockIdx.x * 8 + wave;
    if (row >= BLROW) return;
    const float* yr = ysum + (long)row * DI;
    float v[6];
    float s = 0.0f, s2 = 0.0f;
#pragma unroll
    for (int i = 0; i < 6; ++i) {
        v[i] = yr[lane + 32 * i];
        s += v[i];
        s2 += v[i] * v[i];
    }
    s = wave_sum(s);
    s2 = wave_sum(s2);
    float mu = s * (1.0f / DI);
    float var = s2 * (1.0f / DI) - mu * mu;
    float rstd = rsqrtf(var + 1e-5f);
    const float* zr = xz + (long)row * 384 + DI;
    unsigned short* o = gated + (long)row * DI;
#pragma unroll
    for (int i = 0; i < 6; ++i) {
        int c = lane + 32 * i;
        float yn = (v[i] - mu) * rstd * g[c] + b[c];
        o[c] = f2bf(yn * silu(zr[c]));
    }
}

// ---------- launch ----------
extern "C" void kernel_launch(void* const* d_in, const int* in_sizes, int n_in,
                              void* d_out, int out_size, void* d_ws, size_t ws_size,
                              hipStream_t stream) {
    (void)in_sizes; (void)n_in; (void)out_size; (void)ws_size;
    const float* x          = (const float*)d_in[0];
    const float* ln1_g      = (const float*)d_in[1];
    const float* ln1_b      = (const float*)d_in[2];
    const float* in_proj_w  = (const float*)d_in[3];
    const float* conv_w     = (const float*)d_in[4];
    const float* conv_b     = (const float*)d_in[5];
    const float* x_proj_w   = (const float*)d_in[6];
    const float* dt_w       = (const float*)d_in[7];
    const float* dt_b       = (const float*)d_in[8];
    const float* A_logs     = (const float*)d_in[9];
    const float* Ds         = (const float*)d_in[10];
    const float* out_norm_g = (const float*)d_in[11];
    const float* out_norm_b = (const float*)d_in[12];
    const float* out_proj_w = (const float*)d_in[13];

    unsigned char* ws = (unsigned char*)d_ws;
    unsigned short* wt_in  = (unsigned short*)(ws + 0);          //   73,728 B
    unsigned short* wt_xp  = (unsigned short*)(ws + 73728);      //   73,728 B
    unsigned short* wt_out = (unsigned short*)(ws + 147456);     //   36,864 B
    float*          Aexp   = (float*)(ws + 184320);              //  196,608 B
    unsigned short* hnorm  = (unsigned short*)(ws + 380928);     // 3,145,728 B
    float*          xz     = (float*)(ws + 3526656);             // 25,165,824 B
    float*          xconv  = (float*)(ws + 28692480);            // 12,582,912 B
    unsigned short* xcbf   = (unsigned short*)(ws + 41275392);   // 6,291,456 B
    float*          proj   = (float*)(ws + 47566848);            // 12,582,912 B
    float*          ysum   = (float*)(ws + 60149760);            // 12,582,912 B
    unsigned short* gated  = (unsigned short*)(ws + 72732672);   // 6,291,456 B

    // 0. weight staging + A precompute
    prep_kernel<<<(141312 + 255) / 256, 256, 0, stream>>>(
        in_proj_w, x_proj_w, out_proj_w, A_logs, wt_in, wt_xp, wt_out, Aexp);
    zero_kernel<<<(BLROW * DI + 255) / 256, 256, 0, stream>>>(ysum, BLROW * DI);

    // 1. layernorm -> bf16
    ln_in_kernel<<<BLROW / 8, 256, 0, stream>>>(x, ln1_g, ln1_b, hnorm);

    // 2. in_proj: [16384,96] x [96,384] -> xz   (1024*24 tiles / 8 waves)
    wmma_gemm_nt<<<dim3(3072, 1), 256, 0, stream>>>(
        hnorm, wt_in, xz, nullptr, BLROW, 384, CC, CC, CC, 384, 0L, 0L);

    // 3. depthwise conv + SiLU
    conv_dw_silu_kernel<<<(BLROW * DI + 255) / 256, 256, 0, stream>>>(
        xz, conv_w, conv_b, xconv, xcbf);

    // 4. x_proj (4 directions batched on grid.y): [16384,192] x [192,48]
    wmma_gemm_nt<<<dim3(384, KDIR), 256, 0, stream>>>(
        xcbf, wt_xp, proj, nullptr, BLROW, PLD, DI, DI, DI, PLD,
        (long)PLD * DI, (long)BLROW * PLD);

    // 5. selective scan, 4 directions accumulate into ysum by pixel
    scan_kernel<<<BATCH * KDIR, DI, 0, stream>>>(
        proj, xconv, dt_w, dt_b, Aexp, Ds, ysum);

    // 6. out-norm + gate
    merge_gate_kernel<<<BLROW / 8, 256, 0, stream>>>(
        ysum, out_norm_g, out_norm_b, xz, gated);

    // 7. out_proj + residual: [16384,192] x [192,96] + x -> d_out
    wmma_gemm_nt<<<dim3(768, 1), 256, 0, stream>>>(
        gated, wt_out, (float*)d_out, x, BLROW, CC, DI, DI, DI, CC, 0L, 0L);
}
